// MultiHeadAttentionBlock_51908974739732
// MI455X (gfx1250) — matrix operand order repaired and checked by reading, not compiled, not run on hardware
//
#include <hip/hip_runtime.h>
#include <hip/hip_bf16.h>
#include <math.h>

// ---------------------------------------------------------------------------
// Types for CDNA5 WMMA (wave32): bf16 operands, fp32 accumulators
// ---------------------------------------------------------------------------
typedef __attribute__((ext_vector_type(16))) __bf16 v16bf;
typedef __attribute__((ext_vector_type(8)))  __bf16 v8bf;
typedef __attribute__((ext_vector_type(4)))  __bf16 v4bf;
typedef __attribute__((ext_vector_type(8)))  float  v8f;

union BF16x16 { v16bf v; v8bf h[2]; };

__device__ __forceinline__ v8f wmma_bf16(v16bf a, v16bf b, v8f c) {
    return __builtin_amdgcn_wmma_f32_16x16x32_bf16(
        false, a, false, b, (short)0, c, false, false);
}

// Flat pointers to LDS are {SHARED_BASE[63:32], lds_offset[31:0]} -> low 32
// bits are the wave-relative LDS byte address required by async-to-LDS ops.
__device__ __forceinline__ unsigned lds_u32(const void* p) {
    return (unsigned)(size_t)p;
}

// GLOBAL_LOAD_ASYNC_TO_LDS_B128 (GV mode): VDST = LDS byte addr, VADDR = 64b
// global address. Tracked by ASYNCcnt.
__device__ __forceinline__ void async_copy_b128(unsigned lds, const void* gaddr) {
    asm volatile("global_load_async_to_lds_b128 %0, %1, off"
                 :: "v"(lds), "v"((unsigned long long)(size_t)gaddr)
                 : "memory");
}
__device__ __forceinline__ void wait_async0() { asm volatile("s_wait_asynccnt 0" ::: "memory"); }
__device__ __forceinline__ void wait_async4() { asm volatile("s_wait_asynccnt 4" ::: "memory"); }
__device__ __forceinline__ void wait_async8() { asm volatile("s_wait_asynccnt 8" ::: "memory"); }

// A-frag (16x32 bf16): lane row M = row0+(lane&15); e<8 -> K=k0+hi*8+e,
// e>=8 -> K=k0+16+hi*8+(e-8).  base row-major [rows][ld].
__device__ __forceinline__ v16bf load_a_frag(const __bf16* base, int ld,
                                             int row0, int k0, int hi, int lo) {
    const __bf16* p = base + (size_t)(row0 + lo) * ld + k0 + hi * 8;
    BF16x16 u;
    u.h[0] = *(const v8bf*)(p);
    u.h[1] = *(const v8bf*)(p + 16);
    return u.v;
}
// B-frag (32x16 bf16): lane col N = col0+(lane&15); e<8 -> K=k0+hi*8+e,
// e>=8 -> K=k0+16+hi*8+(e-8).
// base stores B transposed: row-major [N][ld], K contiguous.
__device__ __forceinline__ v16bf load_b_frag(const __bf16* base, int ld,
                                             int col0, int k0, int hi, int lo) {
    const __bf16* p = base + (size_t)(col0 + lo) * ld + k0 + hi * 8;
    BF16x16 u;
    u.h[0] = *(const v8bf*)(p);
    u.h[1] = *(const v8bf*)(p + 16);
    return u.v;
}

// ---------------------------------------------------------------------------
// fp32 -> bf16 convert (vectorized), n4 = elements/4
// ---------------------------------------------------------------------------
__global__ __launch_bounds__(256)
void f32_to_bf16(const float* __restrict__ src, __bf16* __restrict__ dst, int n4) {
    int i = blockIdx.x * 256 + threadIdx.x;
    if (i < n4) {
        float4 f = ((const float4*)src)[i];
        v4bf h = { (__bf16)f.x, (__bf16)f.y, (__bf16)f.z, (__bf16)f.w };
        ((v4bf*)dst)[i] = h;
    }
}

// ---------------------------------------------------------------------------
// GEMM (all-bf16, async double-buffered):
//   C[m,n] = scale * sum_k A[m,k] * W[n,k]
// Block 256 thr = 8 waves; BM=BN=128, BK=32; wave owns 32x64 = 2x4 WMMA tiles.
// STORE_MODE 0: row-major [M][N]. STORE_MODE 1: per-head transposed
//   [B][H][d_k][S] (used for V so attention can async-copy V^T chunks).
// ---------------------------------------------------------------------------
template <bool OUT_IS_BF16, int STORE_MODE>
__global__ __launch_bounds__(256)
void gemm_bf16_wmma(const __bf16* __restrict__ A, const __bf16* __restrict__ W,
                    void* __restrict__ Cp, int M, int N, int K, float scale) {
    constexpr int BM = 128, BN = 128, BK = 32, LDT = BK + 16;  // 48 halfs = 96B rows
    __shared__ __bf16 As[2][BM * LDT];
    __shared__ __bf16 Ws[2][BN * LDT];

    const int tid  = threadIdx.x;
    const int lane = tid & 31, wave = tid >> 5;
    const int hi   = lane >> 4, lo  = lane & 15;
    const int wm   = wave & 3,  wn  = wave >> 2;
    const int m0   = blockIdx.y * BM, n0 = blockIdx.x * BN;

    auto stage = [&](int kt, int buf) {
#pragma unroll
        for (int i = 0; i < 2; ++i) {
            int p = tid + i * 256;                 // 512 v8bf slots per matrix
            int row = p >> 2, col = (p & 3) * 8;
            async_copy_b128(lds_u32(&As[buf][row * LDT + col]),
                            A + (size_t)(m0 + row) * K + (size_t)kt * BK + col);
            async_copy_b128(lds_u32(&Ws[buf][row * LDT + col]),
                            W + (size_t)(n0 + row) * K + (size_t)kt * BK + col);
        }
    };

    v8f zf = {};
    v8f acc[2][4];
#pragma unroll
    for (int i = 0; i < 2; ++i)
#pragma unroll
        for (int j = 0; j < 4; ++j) acc[i][j] = zf;

    const int nk = K / BK;
    stage(0, 0);                                   // prime the pipeline
    for (int kt = 0; kt < nk; ++kt) {
        int cur = kt & 1;
        if (kt + 1 < nk) stage(kt + 1, 1 - cur);   // overlap next copy with math
        if (kt + 2 < nk) {                         // L2 prefetch two tiles ahead
            __builtin_prefetch(A + (size_t)(m0 + (tid >> 1)) * K + (size_t)(kt + 2) * BK, 0, 1);
            __builtin_prefetch(W + (size_t)(n0 + (tid >> 1)) * K + (size_t)(kt + 2) * BK, 0, 1);
        }
        if (kt + 1 < nk) wait_async4(); else wait_async0();
        __syncthreads();

        const __bf16* Ab = &As[cur][0];
        const __bf16* Wb = &Ws[cur][0];
        v16bf a[2], b[4];
#pragma unroll
        for (int mt = 0; mt < 2; ++mt)
            a[mt] = load_a_frag(Ab, LDT, 32 * wm + 16 * mt, 0, hi, lo);
#pragma unroll
        for (int nt = 0; nt < 4; ++nt)
            b[nt] = load_b_frag(Wb, LDT, 64 * wn + 16 * nt, 0, hi, lo);
#pragma unroll
        for (int mt = 0; mt < 2; ++mt)
#pragma unroll
            for (int nt = 0; nt < 4; ++nt)
                acc[mt][nt] = wmma_bf16(a[mt], b[nt], acc[mt][nt]);
        __syncthreads();                            // buffer reuse fence
    }

    // epilogue: D layout M = r + 8*hi, N = lo
#pragma unroll
    for (int mt = 0; mt < 2; ++mt)
#pragma unroll
        for (int nt = 0; nt < 4; ++nt)
#pragma unroll
            for (int r = 0; r < 8; ++r) {
                int m = m0 + 32 * wm + 16 * mt + r + 8 * hi;
                int n = n0 + 64 * wn + 16 * nt + lo;
                float v = acc[mt][nt][r] * scale;
                if constexpr (STORE_MODE == 1) {
                    // m = b*2048+s ; n = h*64+d  ->  [B][H][d][S]
                    int bb = m >> 11, s = m & 2047, h = n >> 6, d = n & 63;
                    ((__bf16*)Cp)[(((size_t)bb * 16 + h) * 64 + d) * 2048 + s] = (__bf16)v;
                } else if constexpr (OUT_IS_BF16) {
                    ((__bf16*)Cp)[(size_t)m * N + n] = (__bf16)v;
                } else {
                    ((float*)Cp)[(size_t)m * N + n] = v;
                }
            }
}

// ---------------------------------------------------------------------------
// Flash attention per (b, h, 64-row q tile); 128 thr = 4 waves, wave w owns
// q rows [16w,16w+16).  K chunks row-major, V chunks pre-transposed [d][kv]
// in global -> both staged with async-to-LDS, double buffered.
// ---------------------------------------------------------------------------
__global__ __launch_bounds__(128)
void attn_wmma(const __bf16* __restrict__ Qb, const __bf16* __restrict__ Kb,
               const __bf16* __restrict__ Vt_g, __bf16* __restrict__ Xb) {
    constexpr int S = 2048, DM = 1024, DK = 64, LDD = DK + 8;   // 72 halfs = 144B
    __shared__ __bf16 Qs[64 * LDD];
    __shared__ __bf16 Ks[2][64 * LDD];
    __shared__ __bf16 Vt[2][64 * LDD];
    __shared__ __bf16 Ps[64 * LDD];

    const int qt = blockIdx.x, bh = blockIdx.y;
    const int bb = bh >> 4, h = bh & 15;
    const int tid  = threadIdx.x;
    const int lane = tid & 31, wave = tid >> 5;
    const int hi   = lane >> 4, lo  = lane & 15;
    const size_t khead = (size_t)bb * S * DM + (size_t)h * DK;          // Q/K rows
    const __bf16* Vh = Vt_g + ((size_t)bb * 16 + h) * (size_t)DK * S;   // V^T head

    auto stage_kv = [&](int j, int buf) {
#pragma unroll
        for (int i = 0; i < 4; ++i) {
            int p = tid + i * 128;                  // 512 v8bf slots each
            int row = p >> 3, col = (p & 7) * 8;
            async_copy_b128(lds_u32(&Ks[buf][row * LDD + col]),
                            Kb + khead + (size_t)(j * 64 + row) * DM + col);
            async_copy_b128(lds_u32(&Vt[buf][row * LDD + col]),
                            Vh + (size_t)row * S + j * 64 + col);
        }
    };

    // async stage Q tile, then prime chunk 0
#pragma unroll
    for (int i = 0; i < 4; ++i) {
        int p = tid + i * 128;
        int row = p >> 3, col = (p & 7) * 8;
        async_copy_b128(lds_u32(&Qs[row * LDD + col]),
                        Qb + khead + (size_t)(qt * 64 + row) * DM + col);
    }
    stage_kv(0, 0);
    wait_async8();                                  // Q (older 4) retired
    __syncthreads();

    v16bf aq[2];
#pragma unroll
    for (int ks = 0; ks < 2; ++ks)
        aq[ks] = load_a_frag(Qs, LDD, 16 * wave, 32 * ks, hi, lo);

    v8f zf = {};
    v8f o[4] = {zf, zf, zf, zf};
    float mrow[8], lrow[8];
#pragma unroll
    for (int r = 0; r < 8; ++r) { mrow[r] = -INFINITY; lrow[r] = 0.0f; }

    for (int j = 0; j < S / 64; ++j) {
        int cur = j & 1;
        if (j + 1 < S / 64) stage_kv(j + 1, 1 - cur);
        if (j + 1 < S / 64) wait_async8(); else wait_async0();
        __syncthreads();

        const __bf16* Kc = &Ks[cur][0];
        const __bf16* Vc = &Vt[cur][0];

        // scores S[16 q x 64 kv] = Q * K^T
        v8f s[4];
#pragma unroll
        for (int nt = 0; nt < 4; ++nt) {
            v8f c = zf;
            c = wmma_bf16(aq[0], load_b_frag(Kc, LDD, 16 * nt, 0,  hi, lo), c);
            c = wmma_bf16(aq[1], load_b_frag(Kc, LDD, 16 * nt, 32, hi, lo), c);
            s[nt] = c;
        }

        // online softmax (row stats across the 16-lane N groups)
#pragma unroll
        for (int r = 0; r < 8; ++r) {
            float mx = fmaxf(fmaxf(s[0][r], s[1][r]), fmaxf(s[2][r], s[3][r]));
#pragma unroll
            for (int off = 8; off >= 1; off >>= 1)
                mx = fmaxf(mx, __shfl_xor(mx, off));
            float mnew  = fmaxf(mrow[r], mx);
            float alpha = __expf(mrow[r] - mnew);
            mrow[r] = mnew;
            float rs = 0.0f;
#pragma unroll
            for (int nt = 0; nt < 4; ++nt) {
                float pv = __expf(s[nt][r] - mnew);
                s[nt][r] = pv;
                rs += pv;
            }
#pragma unroll
            for (int off = 8; off >= 1; off >>= 1)
                rs += __shfl_xor(rs, off);
            lrow[r] = lrow[r] * alpha + rs;
#pragma unroll
            for (int ntd = 0; ntd < 4; ++ntd)
                o[ntd][r] *= alpha;
        }

        // P: D layout -> LDS -> A fragments (wave-local rows; in-wave DS order)
#pragma unroll
        for (int nt = 0; nt < 4; ++nt)
#pragma unroll
            for (int r = 0; r < 8; ++r)
                Ps[(16 * wave + r + 8 * hi) * LDD + 16 * nt + lo] = (__bf16)s[nt][r];

        v16bf ap0 = load_a_frag(Ps, LDD, 16 * wave, 0,  hi, lo);
        v16bf ap1 = load_a_frag(Ps, LDD, 16 * wave, 32, hi, lo);
#pragma unroll
        for (int ntd = 0; ntd < 4; ++ntd) {
            o[ntd] = wmma_bf16(ap0, load_b_frag(Vc, LDD, 16 * ntd, 0,  hi, lo), o[ntd]);
            o[ntd] = wmma_bf16(ap1, load_b_frag(Vc, LDD, 16 * ntd, 32, hi, lo), o[ntd]);
        }
        __syncthreads();                            // buffer reuse fence
    }

    // epilogue: O / l -> Xb[b, q, h*64 + d]
#pragma unroll
    for (int ntd = 0; ntd < 4; ++ntd)
#pragma unroll
        for (int r = 0; r < 8; ++r) {
            int qrow = qt * 64 + 16 * wave + r + 8 * hi;
            int d    = 16 * ntd + lo;
            Xb[khead + (size_t)qrow * DM + d] = (__bf16)(o[ntd][r] / lrow[r]);
        }
}

// ---------------------------------------------------------------------------
// Launch: convert -> 3 projections -> attention -> output projection
// ---------------------------------------------------------------------------
extern "C" void kernel_launch(void* const* d_in, const int* in_sizes, int n_in,
                              void* d_out, int out_size, void* d_ws, size_t ws_size,
                              hipStream_t stream) {
    const float* q  = (const float*)d_in[0];
    const float* k  = (const float*)d_in[1];
    const float* v  = (const float*)d_in[2];
    const float* Wq = (const float*)d_in[3];
    const float* Wk = (const float*)d_in[4];
    const float* Wv = (const float*)d_in[5];
    const float* Wo = (const float*)d_in[6];
    float* out = (float*)d_out;

    const int B = 4, S = 2048, DM = 1024;
    const int M = B * S;                        // 8192
    const size_t PLANE  = (size_t)M * DM;       // activation plane (bf16 elems)
    const size_t WPLANE = (size_t)DM * DM;      // weight plane

    __bf16* qc  = (__bf16*)d_ws;                // reused as attention output Xb
    __bf16* kc  = qc  + PLANE;
    __bf16* vc  = kc  + PLANE;
    __bf16* Wqc = vc  + PLANE;
    __bf16* Wkc = Wqc + WPLANE;
    __bf16* Wvc = Wkc + WPLANE;
    __bf16* Woc = Wvc + WPLANE;
    __bf16* Qb  = Woc + WPLANE;
    __bf16* Kb  = Qb  + PLANE;
    __bf16* Vb  = Kb  + PLANE;                  // [B][H][d_k][S] transposed
    __bf16* Xb  = qc;                           // q plane dead after Q projection

    // fp32 -> bf16 planes
    int n4a = (int)(PLANE / 4), n4w = (int)(WPLANE / 4);
    f32_to_bf16<<<(n4a + 255) / 256, 256, 0, stream>>>(q,  qc,  n4a);
    f32_to_bf16<<<(n4a + 255) / 256, 256, 0, stream>>>(k,  kc,  n4a);
    f32_to_bf16<<<(n4a + 255) / 256, 256, 0, stream>>>(v,  vc,  n4a);
    f32_to_bf16<<<(n4w + 255) / 256, 256, 0, stream>>>(Wq, Wqc, n4w);
    f32_to_bf16<<<(n4w + 255) / 256, 256, 0, stream>>>(Wk, Wkc, n4w);
    f32_to_bf16<<<(n4w + 255) / 256, 256, 0, stream>>>(Wv, Wvc, n4w);
    f32_to_bf16<<<(n4w + 255) / 256, 256, 0, stream>>>(Wo, Woc, n4w);

    dim3 gg(DM / 128, M / 128), bg(256);
    // projections: Q pre-scaled by 1/sqrt(d_k) = 0.125 (exact power of two)
    gemm_bf16_wmma<true, 0><<<gg, bg, 0, stream>>>(qc, Wqc, Qb, M, DM, DM, 0.125f);
    gemm_bf16_wmma<true, 0><<<gg, bg, 0, stream>>>(kc, Wkc, Kb, M, DM, DM, 1.0f);
    gemm_bf16_wmma<true, 1><<<gg, bg, 0, stream>>>(vc, Wvc, Vb, M, DM, DM, 1.0f);
    // attention
    attn_wmma<<<dim3(S / 64, B * 16), dim3(128), 0, stream>>>(Qb, Kb, Vb, Xb);
    // output projection -> fp32 d_out
    gemm_bf16_wmma<false, 0><<<gg, bg, 0, stream>>>(Xb, Woc, out, M, DM, DM, 1.0f);
}
